// Gate_19653770346954
// MI455X (gfx1250) — matrix operand order repaired and checked by reading, not compiled, not run on hardware
//
#include <hip/hip_runtime.h>
#include <math.h>
#include <stdint.h>

typedef _Float16 v16h __attribute__((ext_vector_type(16)));
typedef _Float16 h8  __attribute__((ext_vector_type(8)));
typedef float    v8f __attribute__((ext_vector_type(8)));

// ---------------------------------------------------------------------------
// CDNA5 async global->LDS copy (16B per lane), tracked by ASYNCcnt.
// GV addressing: 64-bit per-lane global address in VGPR pair, LDS offset VGPR.
// ---------------------------------------------------------------------------
__device__ __forceinline__ void async_copy_b128(uint32_t lds_off, const void* gptr) {
  asm volatile("global_load_async_to_lds_b128 %0, %1, off"
               :: "v"(lds_off), "v"((uint64_t)(uintptr_t)gptr)
               : "memory");
}
__device__ __forceinline__ void wait_async_le3() {
  asm volatile("s_wait_asynccnt 0x3" ::: "memory");
}
__device__ __forceinline__ void wait_async_0() {
  asm volatile("s_wait_asynccnt 0x0" ::: "memory");
}

// ---------------------------------------------------------------------------
// im2col for 2x2 stride-2 VALID conv (non-overlapping patches).
// A[m][k], m = (bt*Ho + y)*Wo + x, k = ci*4 + ky*2 + kx ; input NHWC
// ---------------------------------------------------------------------------
template <typename T>
__global__ __launch_bounds__(256)
void im2col_kernel(const T* __restrict__ in, _Float16* __restrict__ A,
                   int H, int Cin, long total) {
  long idx = (long)blockIdx.x * blockDim.x + threadIdx.x;
  if (idx >= total) return;
  int K  = Cin * 4;
  int Wo = H >> 1;
  int k  = (int)(idx % K);
  long m = idx / K;
  int ci  = k >> 2;
  int pos = k & 3;
  int ky = pos >> 1, kx = pos & 1;
  int xo = (int)(m % Wo);
  long r = m / Wo;
  int yo = (int)(r % Wo);
  int bt = (int)(r / Wo);
  long src = (((long)bt * H + (2 * yo + ky)) * H + (2 * xo + kx)) * Cin + ci;
  A[idx] = (_Float16)(float)in[src];
}

// Straight fp32 -> f16 cast (weights are already (N x K) row-major = B^T)
__global__ __launch_bounds__(256)
void cast_to_f16(const float* __restrict__ W, _Float16* __restrict__ B, long total) {
  long idx = (long)blockIdx.x * blockDim.x + threadIdx.x;
  if (idx >= total) return;
  B[idx] = (_Float16)W[idx];
}

// ---------------------------------------------------------------------------
// Tiled WMMA GEMM: C(MxN fp32) = A(MxK f16) * Bt(NxK f16)^T [+ bias]
// 256 threads = 8 waves; block tile 128x64; each wave computes a 32x32 tile
// (2 A fragments x 2 B fragments -> 4 v_wmma per K-step).
// Staging: double-buffered global_load_async_to_lds_b128 (16B/lane, 3 async
// instructions per wave per stage), pipelined with s_wait_asynccnt.
// ---------------------------------------------------------------------------
__global__ __launch_bounds__(256)
void gemm_wmma_f16(const _Float16* __restrict__ A, const _Float16* __restrict__ Bt,
                   float* __restrict__ C, const float* __restrict__ bias,
                   int M, int N, int K) {
  __shared__ __align__(16) _Float16 As[2][128][32];
  __shared__ __align__(16) _Float16 Bs[2][64][32];   // Bs[n][k]

  int tid  = threadIdx.x;
  int lane = tid & 31;
  int wid  = tid >> 5;        // 0..7
  int wm   = wid & 3;         // wave row quarter (32 rows each)
  int wn   = wid >> 2;        // wave col half   (32 cols each)
  long m0  = (long)blockIdx.x * 128;
  long n0  = (long)blockIdx.y * 64;

  // staging: A tile 128x32 halfs = 512 16B-chunks -> 2 per thread;
  //          B tile  64x32 halfs = 256 16B-chunks -> 1 per thread.
  int ar0 = tid >> 2;                 // rows 0..63
  int ar1 = ar0 + 64;                 // rows 64..127
  int ac  = (tid & 3) * 8;
  const _Float16* gA0 = A  + (m0 + ar0) * (long)K + ac;
  const _Float16* gA1 = A  + (m0 + ar1) * (long)K + ac;
  const _Float16* gB  = Bt + (n0 + ar0) * (long)K + ac;
  uint32_t ldsA0[2] = {(uint32_t)(uintptr_t)&As[0][ar0][ac], (uint32_t)(uintptr_t)&As[1][ar0][ac]};
  uint32_t ldsA1[2] = {(uint32_t)(uintptr_t)&As[0][ar1][ac], (uint32_t)(uintptr_t)&As[1][ar1][ac]};
  uint32_t ldsB[2]  = {(uint32_t)(uintptr_t)&Bs[0][ar0][ac], (uint32_t)(uintptr_t)&Bs[1][ar0][ac]};

  // issue stage 0
  async_copy_b128(ldsA0[0], gA0);
  async_copy_b128(ldsA1[0], gA1);
  async_copy_b128(ldsB[0],  gB);

  v8f acc00 = {}, acc01 = {}, acc10 = {}, acc11 = {};
  const int kA = (lane < 16) ? 0 : 8;    // A fragment K base (documented 16-bit layout)
  const int kB = (lane < 16) ? 0 : 8;    // B fragment K base
  const int mr0 = wm * 32 + (lane & 15);
  const int mr1 = mr0 + 16;
  const int nc0 = wn * 32 + (lane & 15);
  const int nc1 = nc0 + 16;

  int nstages = K >> 5;
  for (int s = 0; s < nstages; ++s) {
    int cur = s & 1;
    if (s + 1 < nstages) {
      long koff = (long)(s + 1) * 32;
      async_copy_b128(ldsA0[cur ^ 1], gA0 + koff);
      async_copy_b128(ldsA1[cur ^ 1], gA1 + koff);
      async_copy_b128(ldsB[cur ^ 1],  gB  + koff);
      wait_async_le3();        // in-order: stage s complete, stage s+1 in flight
    } else {
      wait_async_0();
    }
    __syncthreads();

    // A fragments: lane L row M=L&15; elems e<8 -> K=kA+e, e>=8 -> K=kA+16+(e-8)
    h8 a0lo = *(const h8*)&As[cur][mr0][kA];
    h8 a0hi = *(const h8*)&As[cur][mr0][kA + 16];
    h8 a1lo = *(const h8*)&As[cur][mr1][kA];
    h8 a1hi = *(const h8*)&As[cur][mr1][kA + 16];
    // B fragments: lane L col N; elems e<8 -> K=kB+e, e>=8 -> K=kB+16+(e-8)
    h8 bl0 = *(const h8*)&Bs[cur][nc0][kB];
    h8 bh0 = *(const h8*)&Bs[cur][nc0][kB + 16];
    h8 bl1 = *(const h8*)&Bs[cur][nc1][kB];
    h8 bh1 = *(const h8*)&Bs[cur][nc1][kB + 16];
    v16h a0, a1, b0, b1;
#pragma unroll
    for (int e = 0; e < 8; ++e) {
      a0[e] = a0lo[e]; a0[e + 8] = a0hi[e];
      a1[e] = a1lo[e]; a1[e + 8] = a1hi[e];
      b0[e] = bl0[e];  b0[e + 8] = bh0[e];
      b1[e] = bl1[e];  b1[e + 8] = bh1[e];
    }
    acc00 = __builtin_amdgcn_wmma_f32_16x16x32_f16(false, a0, false, b0, (short)0, acc00, false, false);
    acc01 = __builtin_amdgcn_wmma_f32_16x16x32_f16(false, a0, false, b1, (short)0, acc01, false, false);
    acc10 = __builtin_amdgcn_wmma_f32_16x16x32_f16(false, a1, false, b0, (short)0, acc10, false, false);
    acc11 = __builtin_amdgcn_wmma_f32_16x16x32_f16(false, a1, false, b1, (short)0, acc11, false, false);
    __syncthreads();           // protect buffer reuse before next stage's async writes
  }

  // C/D layout: VGPR r -> row r (lanes 0-15) / r+8 (lanes 16-31), col = lane&15
  const int rbase = (lane < 16) ? 0 : 8;
  long col0 = n0 + nc0;
  long col1 = n0 + nc1;
  float badd0 = bias ? bias[col0] : 0.0f;
  float badd1 = bias ? bias[col1] : 0.0f;
#pragma unroll
  for (int r = 0; r < 8; ++r) {
    long row0 = m0 + wm * 32 + rbase + r;
    long row1 = row0 + 16;
    C[row0 * (long)N + col0] = acc00[r] + badd0;
    C[row0 * (long)N + col1] = acc01[r] + badd1;
    C[row1 * (long)N + col0] = acc10[r] + badd0;
    C[row1 * (long)N + col1] = acc11[r] + badd1;
  }
}

// ---------------------------------------------------------------------------
// Per-row: + conv bias, LayerNorm over channels, gamma/beta, exact GELU -> f16
// ---------------------------------------------------------------------------
__global__ __launch_bounds__(256)
void ln_gelu_kernel(const float* __restrict__ C, const float* __restrict__ bias,
                    const float* __restrict__ g, const float* __restrict__ beta,
                    _Float16* __restrict__ out, int N) {
  __shared__ float red[256];
  long row = blockIdx.x;
  const float* rp = C + row * (long)N;
  float v[4];
  int cnt = 0;
  float s = 0.f;
  for (int c = threadIdx.x; c < N; c += 256) {
    float t = rp[c] + bias[c];
    v[cnt++] = t;
    s += t;
  }
  red[threadIdx.x] = s; __syncthreads();
  for (int off = 128; off > 0; off >>= 1) {
    if (threadIdx.x < off) red[threadIdx.x] += red[threadIdx.x + off];
    __syncthreads();
  }
  float mu = red[0] / (float)N;
  __syncthreads();
  float vs = 0.f;
  for (int j = 0; j < cnt; ++j) { float d = v[j] - mu; vs += d * d; }
  red[threadIdx.x] = vs; __syncthreads();
  for (int off = 128; off > 0; off >>= 1) {
    if (threadIdx.x < off) red[threadIdx.x] += red[threadIdx.x + off];
    __syncthreads();
  }
  float rstd = rsqrtf(red[0] / (float)N + 1e-5f);
  cnt = 0;
  for (int c = threadIdx.x; c < N; c += 256) {
    float hn = (v[cnt++] - mu) * rstd;
    float o  = hn * g[c] + beta[c];
    float ge = 0.5f * o * (1.0f + erff(o * 0.70710678118654752f));
    out[row * (long)N + c] = (_Float16)ge;
  }
}

// ---------------------------------------------------------------------------
// Direct ortho-normalized rDFT over T=96 (freqs 1..48), magnitude, mean over C
// ---------------------------------------------------------------------------
__global__ __launch_bounds__(256)
void fft_amp_kernel(const float* __restrict__ h, float* __restrict__ amp) {
  __shared__ float red[256];
  int b = blockIdx.x / 48;
  int f = (blockIdx.x % 48) + 1;
  float w0 = -6.283185307179586f * (float)f / 96.0f;
  float re[4] = {0, 0, 0, 0}, im[4] = {0, 0, 0, 0};
  for (int t = 0; t < 96; ++t) {
    float sn, cs;
    __sincosf(w0 * (float)t, &sn, &cs);
    const float* hp = h + ((long)(b * 96 + t)) * 1024 + threadIdx.x;
#pragma unroll
    for (int j = 0; j < 4; ++j) {
      float val = hp[j * 256];
      re[j] += val * cs;
      im[j] += val * sn;
    }
  }
  float p = 0.f;
#pragma unroll
  for (int j = 0; j < 4; ++j) p += sqrtf(re[j] * re[j] + im[j] * im[j]);
  red[threadIdx.x] = p; __syncthreads();
  for (int off = 128; off > 0; off >>= 1) {
    if (threadIdx.x < off) red[threadIdx.x] += red[threadIdx.x + off];
    __syncthreads();
  }
  if (threadIdx.x == 0)
    amp[b * 48 + (f - 1)] = red[0] * 0.1020620726159658f / 1024.0f; // 1/sqrt(96) / C
}

// ---------------------------------------------------------------------------
// Noisy top-k gate (B=8, n_seg<=32)
// ---------------------------------------------------------------------------
#define MAXSEG 32
__global__ void gate_kernel(const float* __restrict__ amp, const float* __restrict__ w_gate,
                            const float* __restrict__ w_noise, const float* __restrict__ noise,
                            float* __restrict__ out_gates, float* __restrict__ out_load,
                            int n_seg) {
  __shared__ float sm[8][MAXSEG];
  int tid = threadIdx.x;
  const float inv_sqrt2 = 0.70710678118654752f;
  if (tid < 8) {
    int b = tid;
    float clean[MAXSEG], sd[MAXSEG], noisy[MAXSEG], tmp[MAXSEG];
    for (int s = 0; s < n_seg; ++s) {
      float cg = 0.f, cn = 0.f;
      for (int f = 0; f < 48; ++f) {
        float a = amp[b * 48 + f];
        cg += a * w_gate[f * n_seg + s];
        cn += a * w_noise[f * n_seg + s];
      }
      clean[s] = cg;
      float sp = (cn > 20.f) ? cn : log1pf(expf(cn));
      sd[s] = sp + 0.01f;
      noisy[s] = cg + noise[b * n_seg + s] * sd[s];
      tmp[s] = noisy[s];
    }
    float tv[4]; int ti[4];
    for (int j = 0; j < 4; ++j) {
      int best = 0; float bv = -INFINITY;
      for (int s = 0; s < n_seg; ++s)
        if (tmp[s] > bv) { bv = tmp[s]; best = s; }
      tv[j] = bv; ti[j] = best; tmp[best] = -INFINITY;
    }
    float mx = tv[0];
    float e0 = expf(tv[0] - mx), e1 = expf(tv[1] - mx), e2 = expf(tv[2] - mx);
    float den = e0 + e1 + e2;
    for (int s = 0; s < n_seg; ++s) out_gates[b * n_seg + s] = 0.f;
    out_gates[b * n_seg + ti[0]] = e0 / den;
    out_gates[b * n_seg + ti[1]] = e1 / den;
    out_gates[b * n_seg + ti[2]] = e2 / den;
    float thr_in = tv[3], thr_out = tv[2];
    for (int s = 0; s < n_seg; ++s) {
      float z = (noisy[s] > thr_in) ? (clean[s] - thr_in) / sd[s]
                                    : (clean[s] - thr_out) / sd[s];
      sm[b][s] = 0.5f * erfcf(-z * inv_sqrt2);
    }
  }
  __syncthreads();
  if (tid < n_seg) {
    float t = 0.f;
    for (int b = 0; b < 8; ++b) t += sm[b][tid];
    out_load[tid] = t;
  }
}

// ---------------------------------------------------------------------------
extern "C" void kernel_launch(void* const* d_in, const int* in_sizes, int n_in,
                              void* d_out, int out_size, void* d_ws, size_t ws_size,
                              hipStream_t stream) {
  (void)n_in; (void)out_size; (void)ws_size;
  const float* x = (const float*)d_in[0];
  const float *cw[5], *cb[5], *cg[5], *cbeta[5];
  for (int l = 0; l < 5; ++l) {
    cw[l]    = (const float*)d_in[1 + 4 * l + 0];
    cb[l]    = (const float*)d_in[1 + 4 * l + 1];
    cg[l]    = (const float*)d_in[1 + 4 * l + 2];
    cbeta[l] = (const float*)d_in[1 + 4 * l + 3];
  }
  const float* fuse_w  = (const float*)d_in[21];
  const float* fuse_b  = (const float*)d_in[22];
  const float* w_gate  = (const float*)d_in[23];
  const float* w_noise = (const float*)d_in[24];
  const float* noise   = (const float*)d_in[25];
  int n_seg = in_sizes[23] / 48;

  // workspace layout
  unsigned char* W8 = (unsigned char*)d_ws;
  _Float16* Abuf = (_Float16*)(W8 + 0);              // 50.3 MB (layer0 im2col max)
  float*    Cbuf = (float*)(W8 + 50331648ull);       // 50.3 MB (layer0 GEMM out max)
  _Float16* act0 = (_Float16*)(W8 + 100663296ull);   // 25.2 MB
  _Float16* act1 = (_Float16*)(W8 + 125829120ull);   // 12.6 MB
  _Float16* wbuf[6];
  size_t woff = 138412032ull;
  const size_t wsz[6] = {8192, 32768, 131072, 524288, 2097152, 1048576}; // halfs
  for (int i = 0; i < 6; ++i) { wbuf[i] = (_Float16*)(W8 + woff); woff += wsz[i] * 2; }
  float* amp = (float*)(W8 + woff);

  const int Kc[5] = {128, 256, 512, 1024, 2048};
  const int Nc[5] = {64, 128, 256, 512, 1024};
  const int Hs[5] = {32, 16, 8, 4, 2};
  const int Cins[5] = {32, 64, 128, 256, 512};

  // convert weights to f16 (native (N x K) layout is already B^T)
  for (int l = 0; l < 5; ++l) {
    long tot = (long)Nc[l] * Kc[l];
    cast_to_f16<<<(unsigned)((tot + 255) / 256), 256, 0, stream>>>(cw[l], wbuf[l], tot);
  }
  cast_to_f16<<<(1024 * 1024) / 256, 256, 0, stream>>>(fuse_w, wbuf[5], 1024L * 1024L);

  // conv tower: im2col -> WMMA GEMM -> bias+LN+GELU (f16 NHWC activations)
  _Float16* acts[2] = {act0, act1};
  const void* cur_in = (const void*)x;
  for (int l = 0; l < 5; ++l) {
    int H = Hs[l], Cin = Cins[l], Cout = Nc[l];
    int Ho = H >> 1;
    long M = 768L * Ho * Ho;
    int K = Cin * 4;
    long totA = M * (long)K;
    if (l == 0)
      im2col_kernel<float><<<(unsigned)((totA + 255) / 256), 256, 0, stream>>>(
          (const float*)cur_in, Abuf, H, Cin, totA);
    else
      im2col_kernel<_Float16><<<(unsigned)((totA + 255) / 256), 256, 0, stream>>>(
          (const _Float16*)cur_in, Abuf, H, Cin, totA);
    dim3 gg((unsigned)(M / 128), (unsigned)(Cout >= 64 ? Cout / 64 : 1));
    gemm_wmma_f16<<<gg, 256, 0, stream>>>(Abuf, wbuf[l], Cbuf, nullptr, (int)M, Cout, K);
    _Float16* aout = acts[l & 1];
    ln_gelu_kernel<<<(unsigned)M, 256, 0, stream>>>(Cbuf, cb[l], cg[l], cbeta[l], aout, Cout);
    cur_in = aout;
  }

  // fuse: (768 x 1024) = act4 (768 x 1024 f16) * fuse_w^T, + fuse_b
  {
    dim3 gg(768 / 128, 1024 / 64);
    gemm_wmma_f16<<<gg, 256, 0, stream>>>((const _Float16*)cur_in, wbuf[5], Cbuf, fuse_b,
                                          768, 1024, 1024);
  }

  // spectral amplitudes + gate
  fft_amp_kernel<<<8 * 48, 256, 0, stream>>>(Cbuf, amp);
  float* out_gates = (float*)d_out;
  float* out_load  = out_gates + 8 * n_seg;
  gate_kernel<<<1, 32, 0, stream>>>(amp, w_gate, w_noise, noise, out_gates, out_load, n_seg);
}
